// BiLSTMDecoderModel_47107201302637
// MI455X (gfx1250) — compile-verified, matrix operand order checked by reading, not run on hardware
//
#include <hip/hip_runtime.h>
#include <cstdint>
#include <cstddef>

// Problem sizes (fixed by the reference)
#define Bn 256
#define Sn 128
#define En 512
#define Hn 1024
#define Pn 512
#define On 6

typedef __attribute__((ext_vector_type(16))) __bf16 v16bf;
typedef __attribute__((ext_vector_type(8)))  float  v8f;
typedef __attribute__((ext_vector_type(4)))  unsigned int v4u;
typedef __attribute__((ext_vector_type(8)))  int    v8i;
typedef __attribute__((ext_vector_type(4)))  int    v4i;

#define WMMA_BF16(Af, Bf, Cf) \
  __builtin_amdgcn_wmma_f32_16x16x32_bf16(false, (Af), false, (Bf), (short)0, (Cf), false, false)

__device__ __forceinline__ float sigm(float x) { return 1.0f / (1.0f + __expf(-x)); }

// Load a 16x32 bf16 WMMA operand fragment (A or B role) from a row-major
// matrix with leading dimension ldk (elements). ISA layout (CDNA5 ISA):
// lane L: row = L&15 ; lanes 0-15 hold K = k0+0..7 and k0+16..23,
// lanes 16-31 hold K = k0+8..15 and k0+24..31. Each half is one 16B load.
__device__ __forceinline__ v16bf load_frag(const __bf16* __restrict__ base,
                                           int row0, int ldk, int k0) {
  const int lane = threadIdx.x & 31;
  const int r    = lane & 15;
  const int kb   = (lane >> 4) << 3;  // 0 or 8
  const __bf16* p = base + (size_t)(row0 + r) * ldk + (k0 + kb);
  union { v16bf v; uint4 q[2]; } u;
  u.q[0] = *reinterpret_cast<const uint4*>(p);
  u.q[1] = *reinterpret_cast<const uint4*>(p + 16);
  return u.v;
}

// Same fragment layout, but sourced from an LDS-staged 16x32 tile
// (row-major, 32 elements per row).
__device__ __forceinline__ v16bf load_frag_lds(const __bf16* base) {
  const int lane = threadIdx.x & 31;
  const int r    = lane & 15;
  const int kb   = (lane >> 4) << 3;
  const __bf16* p = base + r * 32 + kb;
  union { v16bf v; uint4 q[2]; } u;
  u.q[0] = *reinterpret_cast<const uint4*>(p);
  u.q[1] = *reinterpret_cast<const uint4*>(p + 16);
  return u.v;
}

// -------------------------------------------------------------------------
// Tensor Data Mover: DMA one 16(row) x 32(K) bf16 tile from global memory
// (row stride ldk elements) into LDS at lds_byte_addr. D# layout per
// CDNA5 ISA (group0: count/lds/global/type,
// group1: data_size=2B, tile 32x16, dim0 stride; trailing groups zero =>
// <=2-D tensor). All operands are wave-uniform -> SGPR groups. Tracked by
// TENSORcnt. This toolchain exposes the 6-arg builtin:
//   (v4u g0, v8i g1, v4i, v4i, v8i, i32 cpol)
// -------------------------------------------------------------------------
__device__ __forceinline__ void tdm_load_tile_16x32_bf16(
    unsigned lds_byte_addr, const __bf16* gptr, unsigned ldk_elems) {
  const uint64_t ga = (uint64_t)(uintptr_t)gptr;
  v4u g0;
  g0[0] = 1u;                                   // count=1, user descriptor
  g0[1] = lds_byte_addr;                        // lds_addr (bytes)
  g0[2] = (unsigned)(ga & 0xffffffffu);         // global_addr[31:0]
  g0[3] = (unsigned)((ga >> 32) & 0x1ffffffu)   // global_addr[56:32]
          | (2u << 30);                         // type = 2 ("image")
  v8i g1;
  g1[0] = (int)(1u << 16);                      // wg_mask=0, data_size=1 (2B)
  g1[1] = (int)0xffff0000u;                     // tensor_dim0[15:0]=0xffff (huge)
  g1[2] = (int)0xffff7fffu;                     // tensor_dim0[31:16] | tensor_dim1[15:0]
  g1[3] = (int)((32u << 16) | 0x7fffu);         // tensor_dim1[31:16] | tile_dim0=32
  g1[4] = (int)16u;                             // tile_dim1=16, tile_dim2=0
  g1[5] = (int)ldk_elems;                       // tensor_dim0_stride[31:0]
  g1[6] = 0;                                    // stride[47:32]=0, dim1_stride lo=0
  g1[7] = 0;
  v4i gz4 = {0, 0, 0, 0};                       // groups 2/3: unused (<=2D)
  v8i gz8 = {0, 0, 0, 0, 0, 0, 0, 0};           // trailing group: unused
  __builtin_amdgcn_tensor_load_to_lds(g0, g1, gz4, gz4, gz8, 0);
}

// ---------------------------------------------------------------------------
// Fused LSTM step: gates = x @ Wih^T + h_prev @ Whh^T + bih + bhh ; cell update.
// Block = 8 waves (256 thr). Grid = (H/16 N-tiles, 2 batch halves).
// Wave w owns the 16x16 tile at rows m0 = by*128 + w*16, cols n0.
// The 4 gate weight tiles (shared by all 8 waves) are staged into LDS by the
// Tensor Data Mover each K-chunk; consumers read B fragments via ds_load.
// KH = Hn for normal steps, KH = 0 when h_prev == 0 (backward first step).
// ---------------------------------------------------------------------------
__global__ __launch_bounds__(256) void lstm_step_kernel(
    const __bf16* __restrict__ x,      // (B, E)   bf16
    const __bf16* __restrict__ Wih,    // (4H, E)  bf16
    const __bf16* __restrict__ hprev,  // (B, H)   bf16
    const __bf16* __restrict__ Whh,    // (4H, H)  bf16
    const float*  __restrict__ bih,    // (4H)
    const float*  __restrict__ bhh,    // (4H)
    const float*  __restrict__ cprev,  // (B, H)   f32
    float*        __restrict__ cnew,   // (B, H)   f32 (may alias cprev)
    __bf16*       __restrict__ hnew,   // (B, H)   bf16
    int KH)
{
  __shared__ __attribute__((aligned(16))) __bf16 sB[4 * 16 * 32];  // 4 KB

  const int n0   = blockIdx.x * 16;                 // hidden-column tile
  const int wave = (int)(threadIdx.x >> 5);         // 0..7
  const int m0   = (int)blockIdx.y * 128 + wave * 16;
  const unsigned lds0 = (unsigned)(size_t)(&sB[0]); // LDS byte offset of sB

  v8f acc0 = {}, acc1 = {}, acc2 = {}, acc3 = {};

  // ---- Input contribution: K over E, Wih tiles staged via TDM ----
  for (int k = 0; k < En; k += 32) {
    if (wave == 0) {
#pragma unroll
      for (int g = 0; g < 4; ++g)
        tdm_load_tile_16x32_bf16(lds0 + (unsigned)g * 1024u,
                                 Wih + ((size_t)(g * Hn + n0) * En + k), En);
      __builtin_amdgcn_s_wait_tensorcnt(0);
    }
    __syncthreads();
    v16bf a = load_frag(x, m0, En, k);
    acc0 = WMMA_BF16(a, load_frag_lds(sB + 0 * 512), acc0);
    acc1 = WMMA_BF16(a, load_frag_lds(sB + 1 * 512), acc1);
    acc2 = WMMA_BF16(a, load_frag_lds(sB + 2 * 512), acc2);
    acc3 = WMMA_BF16(a, load_frag_lds(sB + 3 * 512), acc3);
    __syncthreads();
  }

  // ---- Recurrent contribution: K over H (skipped when KH == 0) ----
  for (int k = 0; k < KH; k += 32) {
    if (wave == 0) {
#pragma unroll
      for (int g = 0; g < 4; ++g)
        tdm_load_tile_16x32_bf16(lds0 + (unsigned)g * 1024u,
                                 Whh + ((size_t)(g * Hn + n0) * Hn + k), Hn);
      __builtin_amdgcn_s_wait_tensorcnt(0);
    }
    __syncthreads();
    v16bf a = load_frag(hprev, m0, Hn, k);
    acc0 = WMMA_BF16(a, load_frag_lds(sB + 0 * 512), acc0);
    acc1 = WMMA_BF16(a, load_frag_lds(sB + 1 * 512), acc1);
    acc2 = WMMA_BF16(a, load_frag_lds(sB + 2 * 512), acc2);
    acc3 = WMMA_BF16(a, load_frag_lds(sB + 3 * 512), acc3);
    __syncthreads();
  }

  // Cell update. C/D layout: lane 0-15 -> N=lane, M = m0 + r ;
  // lanes 16-31 -> N = lane-16, M = m0 + 8 + r.
  const int lane = threadIdx.x & 31;
  const int n    = n0 + (lane & 15);
  const int mb   = m0 + ((lane >> 4) << 3);
  const float bi = bih[0 * Hn + n] + bhh[0 * Hn + n];
  const float bf = bih[1 * Hn + n] + bhh[1 * Hn + n];
  const float bg = bih[2 * Hn + n] + bhh[2 * Hn + n];
  const float bo = bih[3 * Hn + n] + bhh[3 * Hn + n];
#pragma unroll
  for (int r = 0; r < 8; ++r) {
    const int m = mb + r;
    const size_t idx = (size_t)m * Hn + n;
    const float gi = acc0[r] + bi;
    const float gf = acc1[r] + bf;
    const float gg = acc2[r] + bg;
    const float go = acc3[r] + bo;
    const float c  = sigm(gf) * cprev[idx] + sigm(gi) * tanhf(gg);
    cnew[idx] = c;
    hnew[idx] = (__bf16)(sigm(go) * tanhf(c));
  }
}

// ---------------------------------------------------------------------------
// Projection: p = concat(hF, hB) @ Wp^T + bp   (B x P), stored bf16.
// Wp is (P, 2H) row-major: cols [0,H) pair with hF, cols [H,2H) with hB.
// Runs once; direct-from-global WMMA is fine here.
// ---------------------------------------------------------------------------
__global__ __launch_bounds__(32) void proj_kernel(
    const __bf16* __restrict__ hF,   // (B, H)
    const __bf16* __restrict__ hB,   // (B, H)
    const __bf16* __restrict__ Wp,   // (P, 2H)
    const float*  __restrict__ bp,   // (P)
    __bf16*       __restrict__ p)    // (B, P)
{
  const int n0 = blockIdx.x * 16;
  const int m0 = blockIdx.y * 16;
  v8f acc = {};
  for (int k = 0; k < Hn; k += 32) {
    v16bf a = load_frag(hF, m0, Hn, k);
    v16bf b = load_frag(Wp, n0, 2 * Hn, k);
    acc = WMMA_BF16(a, b, acc);
  }
  for (int k = 0; k < Hn; k += 32) {
    v16bf a = load_frag(hB, m0, Hn, k);
    v16bf b = load_frag(Wp, n0, 2 * Hn, Hn + k);
    acc = WMMA_BF16(a, b, acc);
  }
  const int lane = threadIdx.x & 31;
  const int n    = n0 + (lane & 15);
  const int mb   = m0 + ((lane >> 4) << 3);
  const float bb = bp[n];
#pragma unroll
  for (int r = 0; r < 8; ++r)
    p[(size_t)(mb + r) * Pn + n] = (__bf16)(acc[r] + bb);
}

// Tiny head: logits = p @ Wc^T + bc ; sigmoid. (256 x 6, K=512) -> plain VALU.
__global__ void head_kernel(const __bf16* __restrict__ p,   // (B, P)
                            const float*  __restrict__ Wc,  // (O, P)
                            const float*  __restrict__ bc,  // (O)
                            float*        __restrict__ out) // (B, O)
{
  const int b = blockIdx.x * blockDim.x + threadIdx.x;
  if (b >= Bn) return;
#pragma unroll
  for (int o = 0; o < On; ++o) {
    float s = bc[o];
    for (int k = 0; k < Pn; ++k)
      s += (float)p[(size_t)b * Pn + k] * Wc[o * Pn + k];
    out[b * On + o] = sigm(s);
  }
}

// Embedding gather + tanh, emit bf16 xs laid out (S, B, E).
__global__ void embed_tanh_kernel(const int*   __restrict__ seq,   // (B, S)
                                  const float* __restrict__ emb,   // (V, E)
                                  __bf16*      __restrict__ xs)    // (S, B, E)
{
  const size_t total = (size_t)Sn * Bn * En;
  size_t i = (size_t)blockIdx.x * blockDim.x + threadIdx.x;
  if (i >= total) return;
  const int    e  = (int)(i % En);
  const size_t sb = i / En;
  const int    b  = (int)(sb % Bn);
  const int    s  = (int)(sb / Bn);
  const int tok = seq[b * Sn + s];
  xs[i] = (__bf16)tanhf(emb[(size_t)tok * En + e]);
}

__global__ void f32_to_bf16_kernel(const float* __restrict__ s,
                                   __bf16* __restrict__ d, int n) {
  const int i = blockIdx.x * blockDim.x + threadIdx.x;
  if (i < n) d[i] = (__bf16)s[i];
}

__global__ void zero_u32_kernel(uint32_t* __restrict__ p, int n) {
  const int i = blockIdx.x * blockDim.x + threadIdx.x;
  if (i < n) p[i] = 0u;
}

// ---------------------------------------------------------------------------
extern "C" void kernel_launch(void* const* d_in, const int* in_sizes, int n_in,
                              void* d_out, int out_size, void* d_ws, size_t ws_size,
                              hipStream_t stream) {
  (void)in_sizes; (void)n_in; (void)out_size; (void)ws_size;
  const int*   seq   = (const int*)  d_in[0];
  const float* embed = (const float*)d_in[1];
  const float* Wih_f = (const float*)d_in[2];
  const float* Whh_f = (const float*)d_in[3];
  const float* bih_f = (const float*)d_in[4];
  const float* bhh_f = (const float*)d_in[5];
  const float* Wih_b = (const float*)d_in[6];
  // d_in[7] = Whh_b: unused (h0 == 0 for the single backward step)
  const float* bih_b = (const float*)d_in[8];
  const float* bhh_b = (const float*)d_in[9];
  const float* Wp    = (const float*)d_in[10];
  const float* bp    = (const float*)d_in[11];
  const float* Wc    = (const float*)d_in[12];
  const float* bc    = (const float*)d_in[13];
  float* out = (float*)d_out;

  char* ws = (char*)d_ws;
  size_t off = 0;
  auto alloc = [&](size_t bytes) -> char* {
    char* q = ws + off;
    off += (bytes + 255) & ~(size_t)255;
    return q;
  };
  __bf16* xs   = (__bf16*)alloc((size_t)Sn * Bn * En * 2);   // 33.6 MB
  __bf16* wihf = (__bf16*)alloc((size_t)4 * Hn * En * 2);    // 4 MB
  __bf16* whhf = (__bf16*)alloc((size_t)4 * Hn * Hn * 2);    // 8 MB
  __bf16* wihb = (__bf16*)alloc((size_t)4 * Hn * En * 2);    // 4 MB
  __bf16* wpb  = (__bf16*)alloc((size_t)Pn * 2 * Hn * 2);    // 2 MB
  __bf16* h0   = (__bf16*)alloc((size_t)Bn * Hn * 2);
  __bf16* h1   = (__bf16*)alloc((size_t)Bn * Hn * 2);
  float*  cf   = (float*) alloc((size_t)Bn * Hn * 4);
  __bf16* hb   = (__bf16*)alloc((size_t)Bn * Hn * 2);
  float*  cb   = (float*) alloc((size_t)Bn * Hn * 4);
  __bf16* pbuf = (__bf16*)alloc((size_t)Bn * Pn * 2);

  const int T = 256;
  auto cvt = [&](const float* s, __bf16* d, int n) {
    f32_to_bf16_kernel<<<(n + T - 1) / T, T, 0, stream>>>(s, d, n);
  };
  cvt(Wih_f, wihf, 4 * Hn * En);
  cvt(Whh_f, whhf, 4 * Hn * Hn);
  cvt(Wih_b, wihb, 4 * Hn * En);
  cvt(Wp,    wpb,  Pn * 2 * Hn);

  {
    const size_t total = (size_t)Sn * Bn * En;
    embed_tanh_kernel<<<(unsigned)((total + T - 1) / T), T, 0, stream>>>(seq, embed, xs);
  }

  // Zero-init state read at t==0 (h0, cf) and the backward cell state (cb).
  zero_u32_kernel<<<((Bn * Hn / 2) + T - 1) / T, T, 0, stream>>>((uint32_t*)h0, Bn * Hn / 2);
  zero_u32_kernel<<<((Bn * Hn)     + T - 1) / T, T, 0, stream>>>((uint32_t*)cf, Bn * Hn);
  zero_u32_kernel<<<((Bn * Hn)     + T - 1) / T, T, 0, stream>>>((uint32_t*)cb, Bn * Hn);

  const dim3 gStep(Hn / 16, 2), bStep(256);

  // Backward LSTM: only b_hs[0] is consumed => one cell on x_0, zero state
  // (KH = 0: Whh loop skipped; hprev never read).
  lstm_step_kernel<<<gStep, bStep, 0, stream>>>(xs, wihb, h0, whhf,
                                                bih_b, bhh_b, cb, cb, hb, 0);

  // Forward scan over S steps, double-buffered h.
  __bf16* hbuf[2] = { h0, h1 };
  for (int t = 0; t < Sn; ++t) {
    lstm_step_kernel<<<gStep, bStep, 0, stream>>>(
        xs + (size_t)t * Bn * En, wihf, hbuf[t & 1], whhf,
        bih_f, bhh_f, cf, cf, hbuf[(t + 1) & 1], Hn);
  }
  __bf16* hF = hbuf[Sn & 1];  // final forward hidden (Sn even -> h0)

  proj_kernel<<<dim3(Pn / 16, Bn / 16), dim3(32), 0, stream>>>(hF, hb, wpb, bp, pbuf);
  head_kernel<<<1, 256, 0, stream>>>(pbuf, Wc, bc, out);
}
